// PAM_Module_17841294147822
// MI455X (gfx1250) — matrix operand order repaired and checked by reading, not compiled, not run on hardware
//
#include <hip/hip_runtime.h>
#include <hip/hip_bf16.h>

// PAM attention for MI455X (gfx1250, wave32). All GEMMs on v_wmma_f32_16x16x32_f16,
// flash-attention streaming so the 268MB N^2 energy matrix is never materialized.

typedef _Float16 h16;
typedef __attribute__((ext_vector_type(16))) _Float16 v16h;
typedef __attribute__((ext_vector_type(8)))  _Float16 v8h;
typedef __attribute__((ext_vector_type(8)))  float    v8f;

#define BB   4
#define CC   512
#define CQKD 64
#define NN   4096

static __device__ __forceinline__ v8f wmma16(v16h a, v16h b, v8f c) {
  return __builtin_amdgcn_wmma_f32_16x16x32_f16(false, a, false, b, (short)0, c, false, false);
}
static __device__ __forceinline__ v16h cat8(v8h lo, v8h hi) {
  return __builtin_shufflevector(lo, hi, 0,1,2,3,4,5,6,7,8,9,10,11,12,13,14,15);
}
// Row-wise reductions across the 16 lanes of a half-wave (matches WMMA C-layout:
// lanes 0-15 hold rows r, lanes 16-31 hold rows r+8). XOR masks < 16 stay in-half.
static __device__ __forceinline__ float rmax16(float x) {
  x = fmaxf(x, __shfl_xor(x, 1, 32));
  x = fmaxf(x, __shfl_xor(x, 2, 32));
  x = fmaxf(x, __shfl_xor(x, 4, 32));
  x = fmaxf(x, __shfl_xor(x, 8, 32));
  return x;
}
static __device__ __forceinline__ float rsum16(float x) {
  x += __shfl_xor(x, 1, 32);
  x += __shfl_xor(x, 2, 32);
  x += __shfl_xor(x, 4, 32);
  x += __shfl_xor(x, 8, 32);
  return x;
}

// ---- Kernel 1: weights f32 -> f16 (layout preserved; [o][c] row-major = B-fragment friendly)
__global__ void pam_wcvt(const float* __restrict__ Wq, const float* __restrict__ Wk,
                         const float* __restrict__ Wv,
                         h16* __restrict__ wqh, h16* __restrict__ wkh, h16* __restrict__ wvh) {
  int i = blockIdx.x * blockDim.x + threadIdx.x;
  if (i < CQKD * CC) { wqh[i] = (h16)Wq[i]; wkh[i] = (h16)Wk[i]; }
  if (i < CC * CC)   { wvh[i] = (h16)Wv[i]; }
}

// ---- Kernel 2: x [B][C][N] f32 -> xh [B][N][C] f16 (LDS tiled transpose, both sides coalesced)
__global__ void pam_xT(const float* __restrict__ x, h16* __restrict__ xh) {
  __shared__ float tile[32][33];
  int b = blockIdx.z;
  int n0 = blockIdx.x * 32, c0 = blockIdx.y * 32;
  int tx = threadIdx.x, ty = threadIdx.y;
#pragma unroll
  for (int i = 0; i < 4; ++i)
    tile[ty + i * 8][tx] = x[((size_t)b * CC + c0 + ty + i * 8) * NN + n0 + tx];
  __syncthreads();
#pragma unroll
  for (int i = 0; i < 4; ++i)
    xh[((size_t)b * NN + n0 + ty + i * 8) * CC + c0 + tx] = (h16)tile[tx][ty + i * 8];
}

// ---- Kernel 3: Q/K projections. Block = (b, 16-pixel tile); wave w -> output channels 16w..16w+15.
// A-fragment (xh rows): lane L holds M=L%16, K = 16*hi + e contiguous -> two 16B loads.
// B-fragment (weight rows): lane needs K = 16*hi + e contiguous -> one 32B load.
__global__ __launch_bounds__(128) void pam_projqk(const h16* __restrict__ xh,
    const h16* __restrict__ wqh, const h16* __restrict__ wkh,
    const float* __restrict__ bq, const float* __restrict__ bk,
    h16* __restrict__ qh, h16* __restrict__ kh) {
  int blk = blockIdx.x;
  int b = blk >> 8;
  int n0 = (blk & 255) << 4;
  int tid = threadIdx.x;
  int wave = tid >> 5, lane = tid & 31;
  int hi = lane >> 4, lm = lane & 15;
  int o0 = wave << 4;
  const h16* xrow  = xh  + (size_t)(b * NN + n0 + lm) * CC;
  const h16* wqrow = wqh + (size_t)(o0 + lm) * CC;
  const h16* wkrow = wkh + (size_t)(o0 + lm) * CC;
  v8f aq = {}; v8f ak = {};
  for (int cb = 0; cb < CC; cb += 32) {
    v16h a = cat8(*(const v8h*)(xrow + cb + hi * 16),
                  *(const v8h*)(xrow + cb + 8 + hi * 16));
    v16h fq = *(const v16h*)(wqrow + cb + hi * 16);
    v16h fk = *(const v16h*)(wkrow + cb + hi * 16);
    aq = wmma16(a, fq, aq);
    ak = wmma16(a, fk, ak);
  }
  float bqv = bq[o0 + lm], bkv = bk[o0 + lm];
#pragma unroll
  for (int r = 0; r < 8; ++r) {           // C-layout: row m = r + 8*hi, col = lm
    int m = n0 + r + hi * 8;
    qh[(size_t)(b * NN + m) * CQKD + o0 + lm] = (h16)(aq[r] + bqv);
    kh[(size_t)(b * NN + m) * CQKD + o0 + lm] = (h16)(ak[r] + bkv);
  }
}

// ---- Kernel 4: V projection, stored TRANSPOSED vT [B][C][N] f16 so that PV B-fragments
// (fixed channel, contiguous kv positions) are single 32B loads in the attention kernel.
__global__ __launch_bounds__(128) void pam_projv(const h16* __restrict__ xh,
    const h16* __restrict__ wvh, const float* __restrict__ bv, h16* __restrict__ vT) {
  int blk = blockIdx.x;
  int b = blk / 2048;
  int rem = blk % 2048;
  int n0 = (rem >> 3) << 4;
  int cg = rem & 7;
  int tid = threadIdx.x;
  int wave = tid >> 5, lane = tid & 31;
  int hi = lane >> 4, lm = lane & 15;
  int c0 = cg * 64 + wave * 16;
  const h16* xrow = xh  + (size_t)(b * NN + n0 + lm) * CC;
  const h16* wrow = wvh + (size_t)(c0 + lm) * CC;
  v8f acc = {};
  for (int cb = 0; cb < CC; cb += 32) {
    v16h a = cat8(*(const v8h*)(xrow + cb + hi * 16),
                  *(const v8h*)(xrow + cb + 8 + hi * 16));
    v16h f = *(const v16h*)(wrow + cb + hi * 16);
    acc = wmma16(a, f, acc);
  }
  float bvv = bv[c0 + lm];
  union { v8h h; uint4 u; } pk;
#pragma unroll
  for (int r = 0; r < 8; ++r) pk.h[r] = (h16)(acc[r] + bvv);  // rows n0+8*hi+r contiguous
  *(uint4*)(vT + (size_t)(b * CC + c0 + lm) * NN + n0 + hi * 8) = pk.u;
}

// ---- Kernel 5: flash attention. Block = (b, 16 query rows); 4 waves x 128 V-channels each.
__global__ __launch_bounds__(128) void pam_attn(const h16* __restrict__ qh,
    const h16* __restrict__ kh, const h16* __restrict__ vT, float* __restrict__ out) {
  __shared__ h16 plds[4][16 * 32];   // per-wave P patch for C->A layout conversion
  int blk = blockIdx.x;
  int b = blk >> 8;
  int n0 = (blk & 255) << 4;
  int tid = threadIdx.x;
  int wave = tid >> 5, lane = tid & 31;
  int hi = lane >> 4, lm = lane & 15;
  int c0 = wave << 7;                // 128 channels per wave

  const h16* qrow = qh + (size_t)(b * NN + n0 + lm) * CQKD;
  v16h a0 = cat8(*(const v8h*)(qrow + hi * 16),      *(const v8h*)(qrow + 8 + hi * 16));
  v16h a1 = cat8(*(const v8h*)(qrow + 32 + hi * 16), *(const v8h*)(qrow + 40 + hi * 16));

  float rm[8], rs[8];
  v8f acc[8];
#pragma unroll
  for (int r = 0; r < 8; ++r) { rm[r] = -1e30f; rs[r] = 0.f; }
#pragma unroll
  for (int ct = 0; ct < 8; ++ct) acc[ct] = (v8f){};

  h16* pw = plds[wave];
  const h16* kbase = kh + (size_t)b * NN * CQKD;
  const h16* vbase = vT + (size_t)b * CC * NN;

  for (int nt = 0; nt < NN; nt += 32) {
    const h16* k0r = kbase + (size_t)(nt + lm) * CQKD;        // key pixels nt..nt+15
    const h16* k1r = kbase + (size_t)(nt + 16 + lm) * CQKD;   // key pixels nt+16..nt+31
    v16h kb00 = *(const v16h*)(k0r + hi * 16);
    v16h kb01 = *(const v16h*)(k0r + 32 + hi * 16);
    v16h kb10 = *(const v16h*)(k1r + hi * 16);
    v16h kb11 = *(const v16h*)(k1r + 32 + hi * 16);
    v8f z = {};
    v8f e0 = wmma16(a0, kb00, z); e0 = wmma16(a1, kb01, e0);
    v8f e1 = wmma16(a0, kb10, z); e1 = wmma16(a1, kb11, e1);

    float p0[8], p1[8], sc[8];
#pragma unroll
    for (int r = 0; r < 8; ++r) {              // online softmax, row = r + 8*hi
      float tm = rmax16(fmaxf(e0[r], e1[r]));
      float nm = fmaxf(rm[r], tm);
      sc[r] = __expf(rm[r] - nm);
      p0[r] = __expf(e0[r] - nm);
      p1[r] = __expf(e1[r] - nm);
      rs[r] = rs[r] * sc[r] + rsum16(p0[r] + p1[r]);
      rm[r] = nm;
    }
#pragma unroll
    for (int ct = 0; ct < 8; ++ct)
#pragma unroll
      for (int r = 0; r < 8; ++r) acc[ct][r] *= sc[r];

    // P (C-layout f32) -> LDS [16 rows][32 kv] f16
#pragma unroll
    for (int r = 0; r < 8; ++r) {
      int ro = (r + hi * 8) * 32;
      pw[ro + lm]      = (h16)p0[r];
      pw[ro + lm + 16] = (h16)p1[r];
    }
    asm volatile("s_wait_dscnt 0" ::: "memory");   // in-wave RAW on LDS
    v16h pa = cat8(*(const v8h*)(pw + lm * 32 + hi * 16),
                   *(const v8h*)(pw + lm * 32 + 8 + hi * 16));

#pragma unroll
    for (int ct = 0; ct < 8; ++ct) {
      v16h vb = *(const v16h*)(vbase + (size_t)(c0 + ct * 16 + lm) * NN + nt + hi * 16);
      acc[ct] = wmma16(pa, vb, acc[ct]);
    }
    asm volatile("" ::: "memory");
  }

  float inv[8];
#pragma unroll
  for (int r = 0; r < 8; ++r) inv[r] = 1.0f / rs[r];
#pragma unroll
  for (int ct = 0; ct < 8; ++ct) {
    float* orow = out + (size_t)(b * CC + c0 + ct * 16 + lm) * NN + n0 + hi * 8;
    float4 w0 = make_float4(acc[ct][0] * inv[0], acc[ct][1] * inv[1],
                            acc[ct][2] * inv[2], acc[ct][3] * inv[3]);
    float4 w1 = make_float4(acc[ct][4] * inv[4], acc[ct][5] * inv[5],
                            acc[ct][6] * inv[6], acc[ct][7] * inv[7]);
    *(float4*)orow       = w0;
    *(float4*)(orow + 4) = w1;
  }
}

extern "C" void kernel_launch(void* const* d_in, const int* in_sizes, int n_in,
                              void* d_out, int out_size, void* d_ws, size_t ws_size,
                              hipStream_t stream) {
  const float* x  = (const float*)d_in[0];
  const float* Wq = (const float*)d_in[1];
  const float* bq = (const float*)d_in[2];
  const float* Wk = (const float*)d_in[3];
  const float* bk = (const float*)d_in[4];
  const float* Wv = (const float*)d_in[5];
  const float* bv = (const float*)d_in[6];
  float* out = (float*)d_out;

  char* ws = (char*)d_ws;
  // workspace layout (bytes), ~36.6 MB total
  h16* xh  = (h16*)(ws + 0);          // B*N*C f16 = 16777216
  h16* vT  = (h16*)(ws + 16777216);   // B*C*N f16 = 16777216
  h16* qh  = (h16*)(ws + 33554432);   // B*N*64 f16 = 2097152
  h16* kh  = (h16*)(ws + 35651584);   // B*N*64 f16 = 2097152
  h16* wqh = (h16*)(ws + 37748736);   // 64*512 f16
  h16* wkh = (h16*)(ws + 37814272);   // 64*512 f16
  h16* wvh = (h16*)(ws + 37879808);   // 512*512 f16

  pam_wcvt<<<(CC * CC + 255) / 256, 256, 0, stream>>>(Wq, Wk, Wv, wqh, wkh, wvh);
  pam_xT<<<dim3(NN / 32, CC / 32, BB), dim3(32, 8), 0, stream>>>(x, xh);
  pam_projqk<<<BB * (NN / 16), 128, 0, stream>>>(xh, wqh, wkh, bq, bk, qh, kh);
  pam_projv<<<BB * (NN / 16) * (CC / 64), 128, 0, stream>>>(xh, wvh, bv, vT);
  pam_attn<<<BB * (NN / 16), 128, 0, stream>>>(qh, kh, vT, out);
}
